// EfficientDETRDecoder_30133490549552
// MI455X (gfx1250) — compile-verified, matrix operand order checked by reading, not run on hardware
//
#include <hip/hip_runtime.h>
#include <hip/hip_bf16.h>
#include <math.h>

// ---------------- problem constants ----------------
constexpr int Bc  = 8;
constexpr int Cc  = 256;
constexpr int Hc  = 100;
constexpr int Wc  = 100;
constexpr int HWc = Hc * Wc;       // 10000
constexpr int Nqc = 300;
constexpr int NHc = 8;
constexpr int NPc = 4;
constexpr int HDc = Cc / NHc;      // 32

typedef __attribute__((ext_vector_type(16))) __bf16 v16bf;
typedef __attribute__((ext_vector_type(8)))  __bf16 v8bf;
typedef __attribute__((ext_vector_type(8)))  float  v8f;

// =====================================================================
// Generic bf16 WMMA GEMM:  out[M][N] = act(A[M][K] * Wt[N][K]^T + bias
//                                           (+ resid[m % res_rows][N]))
// A, Wt bf16 (Wt pre-transposed [N][K]); fp32 accumulate via
// v_wmma_f32_16x16x32_bf16.  One wave per 16-row M tile; N is processed
// in 64-column chunks (4 tiles) so each A fragment feeds 4 WMMAs
// (2.5 b128 loads per WMMA instead of 4), tail handles N%64.
// =====================================================================
__global__ __launch_bounds__(128) void gemm_bf16_wmma(
    const __bf16* __restrict__ A, const __bf16* __restrict__ Wt,
    const float* __restrict__ bias, const float* __restrict__ resid,
    int res_rows, float* __restrict__ outf, __bf16* __restrict__ outb,
    int M, int K, int N, int act)
{
  const int lane   = threadIdx.x & 31;
  const int wid    = blockIdx.x * (blockDim.x >> 5) + (threadIdx.x >> 5);
  const int mtiles = (M + 15) >> 4;
  if (wid >= mtiles) return;                 // wave-uniform exit
  const int m0     = wid << 4;
  const int ntiles = N >> 4;
  const int ksteps = K >> 5;
  const int lhalf  = lane >> 4;              // 0 or 1
  const int l15    = lane & 15;
  const int arow   = m0 + l15;               // A row for this lane (padded alloc)
  const __bf16* aBase = A + (size_t)arow * K + (lhalf << 3);

  // ---- main: 64-column chunks (4 N tiles per A fragment) ----
  const int nchunks = ntiles >> 2;
  for (int ch = 0; ch < nchunks; ++ch) {
    v8f acc[4];
#pragma unroll
    for (int j = 0; j < 4; ++j) acc[j] = (v8f){0.f,0.f,0.f,0.f,0.f,0.f,0.f,0.f};
    const int nb = ch << 6;
    const __bf16* ap = aBase;
    const __bf16* bp = Wt + (size_t)(nb + l15) * K + (lhalf << 3);
    for (int ks = 0; ks < ksteps; ++ks) {
      union { v16bf v; v8bf h[2]; } a;
      a.h[0] = *(const v8bf*)(ap);
      a.h[1] = *(const v8bf*)(ap + 16);
#pragma unroll
      for (int j = 0; j < 4; ++j) {
        union { v16bf v; v8bf h[2]; } b;
        const __bf16* bj = bp + (size_t)(j << 4) * K;
        b.h[0] = *(const v8bf*)(bj);
        b.h[1] = *(const v8bf*)(bj + 16);
        acc[j] = __builtin_amdgcn_wmma_f32_16x16x32_bf16(
            false, a.v, false, b.v, (short)0, acc[j], false, false);
      }
      ap += 32; bp += 32;
    }
#pragma unroll
    for (int j = 0; j < 4; ++j) {
      const int n = nb + (j << 4) + l15;
      const float bv = bias ? bias[n] : 0.0f;
#pragma unroll
      for (int r = 0; r < 8; ++r) {
        const int m = m0 + r + (lhalf << 3);
        if (m < M) {
          float v = acc[j][r] + bv;
          if (resid) v += resid[(size_t)(m % res_rows) * N + n];
          if (act == 1) v = v > 0.f ? v : 0.f;
          const size_t o = (size_t)m * N + n;
          if (outf) outf[o] = v;
          if (outb) outb[o] = (__bf16)v;
        }
      }
    }
  }

  // ---- tail: remaining N tiles one at a time ----
  for (int nt = nchunks << 2; nt < ntiles; ++nt) {
    v8f acc = {0.f,0.f,0.f,0.f,0.f,0.f,0.f,0.f};
    const int n = (nt << 4) + l15;
    const __bf16* ap = aBase;
    const __bf16* bp = Wt + (size_t)n * K + (lhalf << 3);
    for (int ks = 0; ks < ksteps; ++ks) {
      union { v16bf v; v8bf h[2]; } a, b;
      a.h[0] = *(const v8bf*)(ap);
      a.h[1] = *(const v8bf*)(ap + 16);
      b.h[0] = *(const v8bf*)(bp);
      b.h[1] = *(const v8bf*)(bp + 16);
      acc = __builtin_amdgcn_wmma_f32_16x16x32_bf16(
          false, a.v, false, b.v, (short)0, acc, false, false);
      ap += 32; bp += 32;
    }
    const float bv = bias ? bias[n] : 0.0f;
#pragma unroll
    for (int r = 0; r < 8; ++r) {
      const int m = m0 + r + (lhalf << 3);
      if (m < M) {
        float v = acc[r] + bv;
        if (resid) v += resid[(size_t)(m % res_rows) * N + n];
        if (act == 1) v = v > 0.f ? v : 0.f;
        const size_t o = (size_t)m * N + n;
        if (outf) outf[o] = v;
        if (outb) outb[o] = (__bf16)v;
      }
    }
  }
}

// ---------- weight slice: fp32 [K][Ntot] cols[col0:col0+ncols] -> bf16 [ncols][K]
__global__ void wt_prep(const float* __restrict__ Wf, __bf16* __restrict__ Wt,
                        int K, int Ntot, int col0, int ncols)
{
  int t = blockIdx.x * blockDim.x + threadIdx.x;
  if (t >= ncols * K) return;
  int k = t % K, n = t / K;
  Wt[t] = (__bf16)Wf[(size_t)k * Ntot + col0 + n];
}

// ---------- LayerNorm over rows of [rows][C] fp32 -> bf16, one wave/row
__global__ __launch_bounds__(256) void ln_rows(
    const float* __restrict__ x, const float* __restrict__ g,
    const float* __restrict__ bta, __bf16* __restrict__ ybf, int rows, int C)
{
  const int lane = threadIdx.x & 31;
  const int row  = blockIdx.x * (blockDim.x >> 5) + (threadIdx.x >> 5);
  if (row >= rows) return;
  const float* xr = x + (size_t)row * C;
  float s = 0.f, s2 = 0.f;
  for (int c = lane; c < C; c += 32) { float v = xr[c]; s += v; s2 += v * v; }
  for (int o = 16; o; o >>= 1) { s += __shfl_xor(s, o, 32); s2 += __shfl_xor(s2, o, 32); }
  const float mean = s / C;
  const float inv  = rsqrtf(s2 / C - mean * mean + 1e-5f);
  __bf16* yr = ybf + (size_t)row * C;
  for (int c = lane; c < C; c += 32)
    yr[c] = (__bf16)((xr[c] - mean) * inv * g[c] + bta[c]);
}

// ---------- reference points: sigmoid(query_embed @ ref_w + ref_b)
__global__ void ref_kernel(const float* __restrict__ qe, const float* __restrict__ rw,
                           const float* __restrict__ rb, float* __restrict__ ref)
{
  int t = blockIdx.x * blockDim.x + threadIdx.x;
  if (t >= Nqc * 2) return;
  int q = t >> 1, j = t & 1;
  float s = rb[j];
  for (int c = 0; c < Cc; ++c) s += qe[(size_t)q * Cc + c] * rw[c * 2 + j];
  ref[t] = 1.f / (1.f + __expf(-s));
}

// ---------- self-attention scores (scaled), thread per (h,qi,kj)
__global__ void sa_scores(const float* __restrict__ qb, const float* __restrict__ kb,
                          float* __restrict__ sc)
{
  int t = blockIdx.x * blockDim.x + threadIdx.x;
  if (t >= NHc * Nqc * Nqc) return;
  int kj = t % Nqc; int rest = t / Nqc; int qi = rest % Nqc; int h = rest / Nqc;
  const float* qr = qb + (size_t)qi * Cc + h * HDc;
  const float* kr = kb + (size_t)kj * Cc + h * HDc;
  float s = 0.f;
  for (int d = 0; d < HDc; ++d) s += qr[d] * kr[d];
  sc[t] = s * 0.1767766952966369f;   // 1/sqrt(32)
}

// ---------- row softmax (in place), one wave per row
__global__ __launch_bounds__(256) void softmax_rows(float* __restrict__ sc,
                                                    int rows, int ncol)
{
  int lane = threadIdx.x & 31;
  int row  = blockIdx.x * (blockDim.x >> 5) + (threadIdx.x >> 5);
  if (row >= rows) return;
  float* p = sc + (size_t)row * ncol;
  float mx = -1e30f;
  for (int j = lane; j < ncol; j += 32) mx = fmaxf(mx, p[j]);
  for (int o = 16; o; o >>= 1) mx = fmaxf(mx, __shfl_xor(mx, o, 32));
  float s = 0.f;
  for (int j = lane; j < ncol; j += 32) { float e = __expf(p[j] - mx); p[j] = e; s += e; }
  for (int o = 16; o; o >>= 1) s += __shfl_xor(s, o, 32);
  float inv = 1.f / s;
  for (int j = lane; j < ncol; j += 32) p[j] *= inv;
}

// ---------- attn @ V, thread per (qi, c) -> bf16
__global__ void sa_apply(const float* __restrict__ sc, const float* __restrict__ vb,
                         __bf16* __restrict__ obf)
{
  int t = blockIdx.x * blockDim.x + threadIdx.x;
  if (t >= Nqc * Cc) return;
  int c = t % Cc, qi = t / Cc;
  int h = c / HDc;
  const float* pr = sc + ((size_t)h * Nqc + qi) * Nqc;
  float s = 0.f;
  for (int kj = 0; kj < Nqc; ++kj) s += pr[kj] * vb[(size_t)kj * Cc + c];
  obf[(size_t)qi * Cc + c] = (__bf16)s;
}

// ---------- sampling locations + point-softmax weights, thread per (q,h)
__global__ void loc_kernel(const float* __restrict__ off, const float* __restrict__ awr,
                           const float* __restrict__ ref, float* __restrict__ loc,
                           float* __restrict__ aws)
{
  int t = blockIdx.x * blockDim.x + threadIdx.x;
  if (t >= Nqc * NHc) return;
  int h = t % NHc, q = t / NHc;
  float a[NPc]; float mx = -1e30f;
  for (int p = 0; p < NPc; ++p) {
    a[p] = awr[(size_t)q * (NHc * NPc) + h * NPc + p];
    mx = fmaxf(mx, a[p]);
  }
  float s = 0.f;
  for (int p = 0; p < NPc; ++p) { a[p] = __expf(a[p] - mx); s += a[p]; }
  float rx = ref[q * 2 + 0], ry = ref[q * 2 + 1];
  for (int p = 0; p < NPc; ++p) {
    float ox = off[(size_t)q * (NHc * NPc * 2) + (h * NPc + p) * 2 + 0];
    float oy = off[(size_t)q * (NHc * NPc * 2) + (h * NPc + p) * 2 + 1];
    float lx = fminf(fmaxf(rx + ox / (float)Wc * 0.1f, 0.f), 1.f);
    float ly = fminf(fmaxf(ry + oy / (float)Hc * 0.1f, 0.f), 1.f);
    size_t o = ((size_t)t * NPc + p) * 2;
    loc[o] = lx; loc[o + 1] = ly;
    aws[(size_t)t * NPc + p] = a[p] / s;
  }
}

// ---------- fused sine-pos + NCHW->NHWC transpose -> bf16 activation
__global__ __launch_bounds__(256) void feat_prep(const float* __restrict__ fm,
                                                 __bf16* __restrict__ Abf)
{
  __shared__ float tile[32][33];
  __shared__ float invt[32];
  const int b = blockIdx.z, ctile = blockIdx.y, hwt = blockIdx.x;
  const int t = threadIdx.x;
  if (t < 32) {
    int c  = ctile * 32 + t;
    int cc = c & 127;                          // channel within y/x half
    invt[t] = __powf(10000.0f, -((float)(cc & ~1) / 128.0f));
  }
  __syncthreads();
  const int lane = t & 31, r = t >> 5;
#pragma unroll
  for (int p = 0; p < 4; ++p) {
    int crow = r + p * 8;
    int c  = ctile * 32 + crow;
    int hw = hwt * 32 + lane;
    float v = 0.f;
    if (hw < HWc) {
      int hh = hw / Wc, ww = hw % Wc;
      float coord = (c < 128) ? ((float)hh / Hc) : ((float)ww / Wc);
      float arg = coord * invt[crow];
      float pos = (c & 1) ? __cosf(arg) : __sinf(arg);
      v = fm[((size_t)b * Cc + c) * HWc + hw] + pos;
    }
    tile[crow][lane] = v;
  }
  __syncthreads();
#pragma unroll
  for (int p = 0; p < 4; ++p) {
    int hwrow = r + p * 8;
    int hw = hwt * 32 + hwrow;
    int c  = ctile * 32 + lane;
    if (hw < HWc)
      Abf[((size_t)b * HWc + hw) * Cc + c] = (__bf16)tile[lane][hwrow];
  }
}

// ---------- deformable bilinear gather: wave per (b,q,h), lane = channel d
__global__ __launch_bounds__(256) void deform_gather(const __bf16* __restrict__ val,
    const float* __restrict__ loc, const float* __restrict__ aws,
    __bf16* __restrict__ sbf)
{
  int lane = threadIdx.x & 31;
  int wid  = blockIdx.x * (blockDim.x >> 5) + (threadIdx.x >> 5);
  if (wid >= Bc * Nqc * NHc) return;
  int h = wid % NHc; int rest = wid / NHc; int q = rest % Nqc; int b = rest / Nqc;
  const __bf16* vb = val + (size_t)b * HWc * Cc + h * HDc + lane;
  float acc = 0.f;
  for (int p = 0; p < NPc; ++p) {
    size_t lo = ((size_t)(q * NHc + h) * NPc + p) * 2;
    float lx = loc[lo], ly = loc[lo + 1];
    float a  = aws[(size_t)(q * NHc + h) * NPc + p];
    float gx = lx * (float)Wc - 0.5f;          // align_corners=False mapping
    float gy = ly * (float)Hc - 0.5f;
    float x0 = floorf(gx), y0 = floorf(gy);
    float wx = gx - x0, wy = gy - y0;
    int x0i = min(max((int)x0, 0), Wc - 1);
    int x1i = min(max((int)x0 + 1, 0), Wc - 1);
    int y0i = min(max((int)y0, 0), Hc - 1);
    int y1i = min(max((int)y0 + 1, 0), Hc - 1);
    float v00 = (float)vb[(size_t)(y0i * Wc + x0i) * Cc];
    float v01 = (float)vb[(size_t)(y0i * Wc + x1i) * Cc];
    float v10 = (float)vb[(size_t)(y1i * Wc + x0i) * Cc];
    float v11 = (float)vb[(size_t)(y1i * Wc + x1i) * Cc];
    acc += a * (v00 * (1.f - wx) * (1.f - wy) + v01 * wx * (1.f - wy)
              + v10 * (1.f - wx) * wy        + v11 * wx * wy);
  }
  sbf[((size_t)(b * Nqc + q)) * Cc + h * HDc + lane] = (__bf16)acc;
}

// ---------- output heads: bbox (sigmoid) + objectness
__global__ void heads_kernel(const __bf16* __restrict__ bb, const float* __restrict__ w2,
    const float* __restrict__ b2, const float* __restrict__ q3,
    const float* __restrict__ ow, const float* __restrict__ ob, float* __restrict__ out)
{
  int t = blockIdx.x * blockDim.x + threadIdx.x;
  if (t >= Bc * Nqc * 5) return;
  int j = t % 5, m = t / 5;
  if (j < 4) {
    float s = b2[j];
    for (int c = 0; c < Cc; ++c) s += (float)bb[(size_t)m * Cc + c] * w2[c * 4 + j];
    out[(size_t)m * 4 + j] = 1.f / (1.f + __expf(-s));
  } else {
    float s = ob[0];
    for (int c = 0; c < Cc; ++c) s += q3[(size_t)m * Cc + c] * ow[c];
    out[(size_t)Bc * Nqc * 4 + m] = s;
  }
}

// =====================================================================
static inline size_t aln(size_t x) { return (x + 255) & ~(size_t)255; }

extern "C" void kernel_launch(void* const* d_in, const int* in_sizes, int n_in,
                              void* d_out, int out_size, void* d_ws, size_t ws_size,
                              hipStream_t stream)
{
  const float* feat_map = (const float*)d_in[0];
  const float* query_e  = (const float*)d_in[1];
  const float* ref_w    = (const float*)d_in[2];
  const float* ref_b    = (const float*)d_in[3];
  const float* sa_in_w  = (const float*)d_in[4];
  const float* sa_in_b  = (const float*)d_in[5];
  const float* sa_out_w = (const float*)d_in[6];
  const float* sa_out_b = (const float*)d_in[7];
  const float* off_w    = (const float*)d_in[8];
  const float* off_b    = (const float*)d_in[9];
  const float* attw_w   = (const float*)d_in[10];
  const float* attw_b   = (const float*)d_in[11];
  const float* val_w    = (const float*)d_in[12];
  const float* val_b    = (const float*)d_in[13];
  const float* ca_out_w = (const float*)d_in[14];
  const float* ca_out_b = (const float*)d_in[15];
  const float* ffn_w1   = (const float*)d_in[16];
  const float* ffn_b1   = (const float*)d_in[17];
  const float* ffn_w2   = (const float*)d_in[18];
  const float* ffn_b2   = (const float*)d_in[19];
  const float* ln1_g    = (const float*)d_in[20];
  const float* ln1_b    = (const float*)d_in[21];
  const float* ln2_g    = (const float*)d_in[22];
  const float* ln2_b    = (const float*)d_in[23];
  const float* ln3_g    = (const float*)d_in[24];
  const float* ln3_b    = (const float*)d_in[25];
  const float* bbox_w1  = (const float*)d_in[26];
  const float* bbox_b1  = (const float*)d_in[27];
  const float* bbox_w2  = (const float*)d_in[28];
  const float* bbox_b2  = (const float*)d_in[29];
  const float* obj_w    = (const float*)d_in[30];
  const float* obj_b    = (const float*)d_in[31];
  float* out = (float*)d_out;

  // --------- workspace bump allocator ---------
  char* ws = (char*)d_ws;
  size_t off = 0;
  auto alloc = [&](size_t bytes) { char* p = ws + off; off += aln(bytes); return p; };

  const int Mq  = Nqc;                 // 300
  const int Mqp = 304;                 // padded to 16
  const int Mbq = Bc * Nqc;            // 2400 (mult of 16)
  const int Mv  = Bc * HWc;            // 80000 (mult of 16)

  __bf16* featbf = (__bf16*)alloc((size_t)Mv * Cc * 2);
  __bf16* valbf  = (__bf16*)alloc((size_t)Mv * Cc * 2);
  __bf16* WtQ  = (__bf16*)alloc(256 * 256 * 2);
  __bf16* WtK  = (__bf16*)alloc(256 * 256 * 2);
  __bf16* WtV  = (__bf16*)alloc(256 * 256 * 2);
  __bf16* WtSO = (__bf16*)alloc(256 * 256 * 2);
  __bf16* WtOF = (__bf16*)alloc(64 * 256 * 2);
  __bf16* WtAW = (__bf16*)alloc(32 * 256 * 2);
  __bf16* WtVL = (__bf16*)alloc(256 * 256 * 2);
  __bf16* WtCA = (__bf16*)alloc(256 * 256 * 2);
  __bf16* WtF1 = (__bf16*)alloc(512 * 256 * 2);
  __bf16* WtF2 = (__bf16*)alloc(256 * 512 * 2);
  __bf16* WtB1 = (__bf16*)alloc(256 * 256 * 2);
  __bf16* qn1  = (__bf16*)alloc((size_t)Mqp * Cc * 2);
  float*  qf   = (float*) alloc((size_t)Mq * Cc * 4);
  float*  kf   = (float*) alloc((size_t)Mq * Cc * 4);
  float*  vf   = (float*) alloc((size_t)Mq * Cc * 4);
  float*  sc   = (float*) alloc((size_t)NHc * Mq * Mq * 4);
  __bf16* obf  = (__bf16*)alloc((size_t)Mqp * Cc * 2);
  float*  q1   = (float*) alloc((size_t)Mq * Cc * 4);
  __bf16* qn2  = (__bf16*)alloc((size_t)Mqp * Cc * 2);
  float*  offf = (float*) alloc((size_t)Mq * 64 * 4);
  float*  awr  = (float*) alloc((size_t)Mq * 32 * 4);
  float*  reff = (float*) alloc((size_t)Mq * 2 * 4);
  float*  locf = (float*) alloc((size_t)Mq * NHc * NPc * 2 * 4);
  float*  awsf = (float*) alloc((size_t)Mq * NHc * NPc * 4);
  __bf16* sbf  = (__bf16*)alloc((size_t)Mbq * Cc * 2);
  float*  q2   = (float*) alloc((size_t)Mbq * Cc * 4);
  __bf16* qn3  = (__bf16*)alloc((size_t)Mbq * Cc * 2);
  __bf16* fh   = (__bf16*)alloc((size_t)Mbq * 512 * 2);
  float*  q3   = (float*) alloc((size_t)Mbq * Cc * 4);
  __bf16* q3b  = (__bf16*)alloc((size_t)Mbq * Cc * 2);
  __bf16* bbh  = (__bf16*)alloc((size_t)Mbq * Cc * 2);
  (void)ws_size; (void)in_sizes; (void)n_in; (void)out_size;

  auto gemm = [&](const __bf16* A, const __bf16* Wt, const float* bias,
                  const float* resid, int res_rows, float* of, __bf16* ob,
                  int M, int K, int N, int act) {
    int mtiles = (M + 15) / 16;
    int grid = (mtiles + 3) / 4;                // 4 waves / block of 128
    gemm_bf16_wmma<<<grid, 128, 0, stream>>>(A, Wt, bias, resid, res_rows,
                                             of, ob, M, K, N, act);
  };
  auto prep = [&](const float* Wf, __bf16* Wt, int K, int Ntot, int c0, int nc) {
    int n = nc * K;
    wt_prep<<<(n + 255) / 256, 256, 0, stream>>>(Wf, Wt, K, Ntot, c0, nc);
  };

  // ---- weight transposes to bf16 [N][K]
  prep(sa_in_w, WtQ, 256, 768, 0, 256);
  prep(sa_in_w, WtK, 256, 768, 256, 256);
  prep(sa_in_w, WtV, 256, 768, 512, 256);
  prep(sa_out_w, WtSO, 256, 256, 0, 256);
  prep(off_w,  WtOF, 256, 64, 0, 64);
  prep(attw_w, WtAW, 256, 32, 0, 32);
  prep(val_w,  WtVL, 256, 256, 0, 256);
  prep(ca_out_w, WtCA, 256, 256, 0, 256);
  prep(ffn_w1, WtF1, 256, 512, 0, 512);
  prep(ffn_w2, WtF2, 512, 256, 0, 256);
  prep(bbox_w1, WtB1, 256, 256, 0, 256);

  // ---- batch-independent query path
  ref_kernel<<<(Nqc * 2 + 255) / 256, 256, 0, stream>>>(query_e, ref_w, ref_b, reff);
  ln_rows<<<(Mq + 7) / 8, 256, 0, stream>>>(query_e, ln1_g, ln1_b, qn1, Mq, Cc);
  gemm(qn1, WtQ, sa_in_b,       nullptr, Mq, qf, nullptr, Mq, 256, 256, 0);
  gemm(qn1, WtK, sa_in_b + 256, nullptr, Mq, kf, nullptr, Mq, 256, 256, 0);
  gemm(qn1, WtV, sa_in_b + 512, nullptr, Mq, vf, nullptr, Mq, 256, 256, 0);
  sa_scores<<<(NHc * Mq * Mq + 255) / 256, 256, 0, stream>>>(qf, kf, sc);
  softmax_rows<<<(NHc * Mq + 7) / 8, 256, 0, stream>>>(sc, NHc * Mq, Mq);
  sa_apply<<<(Mq * Cc + 255) / 256, 256, 0, stream>>>(sc, vf, obf);
  gemm(obf, WtSO, sa_out_b, query_e, Mq, q1, nullptr, Mq, 256, 256, 0);
  ln_rows<<<(Mq + 7) / 8, 256, 0, stream>>>(q1, ln2_g, ln2_b, qn2, Mq, Cc);
  gemm(qn2, WtOF, off_b,  nullptr, Mq, offf, nullptr, Mq, 256, 64, 0);
  gemm(qn2, WtAW, attw_b, nullptr, Mq, awr,  nullptr, Mq, 256, 32, 0);
  loc_kernel<<<(Mq * NHc + 255) / 256, 256, 0, stream>>>(offf, awr, reff, locf, awsf);

  // ---- batch path: pos-embed + transpose, value GEMM, gather, CA out
  dim3 fg((HWc + 31) / 32, Cc / 32, Bc);
  feat_prep<<<fg, 256, 0, stream>>>(feat_map, featbf);
  gemm(featbf, WtVL, val_b, nullptr, Mv, nullptr, valbf, Mv, 256, 256, 0);
  deform_gather<<<(Bc * Nqc * NHc + 7) / 8, 256, 0, stream>>>(valbf, locf, awsf, sbf);
  gemm(sbf, WtCA, ca_out_b, q1, Mq, q2, nullptr, Mbq, 256, 256, 0);

  // ---- FFN + heads
  ln_rows<<<(Mbq + 7) / 8, 256, 0, stream>>>(q2, ln3_g, ln3_b, qn3, Mbq, Cc);
  gemm(qn3, WtF1, ffn_b1, nullptr, Mbq, nullptr, fh, Mbq, 256, 512, 1);
  gemm(fh, WtF2, ffn_b2, q2, Mbq, q3, q3b, Mbq, 512, 256, 0);
  gemm(q3b, WtB1, bbox_b1, nullptr, Mbq, nullptr, bbh, Mbq, 256, 256, 1);
  heads_kernel<<<(Bc * Nqc * 5 + 255) / 256, 256, 0, stream>>>(
      bbh, bbox_w2, bbox_b2, q3, obj_w, obj_b, out);
}
